// Attention2_63660005261556
// MI455X (gfx1250) — compile-verified, matrix operand order checked by reading, not run on hardware
//
#include <hip/hip_runtime.h>

typedef __attribute__((ext_vector_type(16))) __bf16 v16bf;
typedef __attribute__((ext_vector_type(8)))  __bf16 v8bf;
typedef __attribute__((ext_vector_type(8)))  float  v8f;
typedef __attribute__((ext_vector_type(4)))  float  f4;

#define BATCH 256
#define LSEQ  2048
#define DIM   128
#define TILE_M 32   // two 16-row WMMA sub-tiles per iteration
#define NSUB  2
#define XPAD  4     // fp32 tile pad (floats): row stride 132
#define BPAD  8     // bf16 tile pad (elems):  row stride 272B
#define RPAD  17    // red_sh row stride (floats, odd -> conflict-free)

// gfx1250 hardware tanh transcendental (V_TANH_F32) — probe-confirmed
#if defined(__has_builtin)
#  if __has_builtin(__builtin_amdgcn_tanhf)
#    define FAST_TANH(x) __builtin_amdgcn_tanhf(x)
#  endif
#endif
#ifndef FAST_TANH
#  define FAST_TANH(x) tanhf(x)
#endif

__global__ __launch_bounds__(256) void attn_additive_kernel(
    const float* __restrict__ inputs,  // [B, L, DIM]
    const float* __restrict__ query,   // [B, DIM]
    const float* __restrict__ mask,    // [B, L]
    const float* __restrict__ Ww,      // [DIM, DIM]  (out, in)
    const float* __restrict__ Wb,      // [DIM]
    const float* __restrict__ Uw,      // [DIM, DIM]
    const float* __restrict__ Ub,      // [DIM]
    const float* __restrict__ Vw,      // [DIM]
    float* __restrict__ out)           // [B, DIM]
{
    __shared__ float  u_sh[DIM];
    __shared__ float  x_sh[2][TILE_M][DIM + XPAD];   // fp32 tiles, double-buffered
    __shared__ __bf16 xb_sh[2][TILE_M][DIM + BPAD];  // bf16 tiles, double-buffered
    __shared__ float  part_sh[TILE_M][DIM + XPAD];   // weighted tanh values
    __shared__ float  red_sh[TILE_M][RPAD];          // 16 partials per row
    __shared__ float  e_sh[TILE_M];
    __shared__ float  num_sh[2][DIM];

    const int b    = blockIdx.x;
    const int tid  = threadIdx.x;
    const int wave = tid >> 5;        // 0..7 : which 16-col slice of OUT_DIM
    const int lane = tid & 31;
    const int lhalf = lane >> 4;      // 0/1 : K sub-block / row-half selector
    const int l15   = lane & 15;      // row (A) / column-of-Ww (B)

    const float* __restrict__ Xb = inputs + (size_t)b * LSEQ * DIM;

    // ---- u = Uw @ query[b] + Ub  (tiny) ----
    if (tid < DIM) {
        const float* q  = query + b * DIM;
        const float* ur = Uw + tid * DIM;
        float acc = Ub[tid];
        #pragma unroll 8
        for (int i = 0; i < DIM; ++i) acc = fmaf(ur[i], q[i], acc);
        u_sh[tid] = acc;
    }
    __syncthreads();

    // ---- loop-invariant B fragments: B[k,n] = Ww[wave*16 + n, k] ----
    v16bf bfrag[4];
    {
        const float* wr = Ww + (wave * 16 + l15) * DIM;
        #pragma unroll
        for (int kf = 0; kf < 4; ++kf) {
            const int k0 = kf * 32 + lhalf * 8;
            #pragma unroll
            for (int j = 0; j < 8; ++j) {
                bfrag[kf][j]     = (__bf16)wr[k0 + j];
                bfrag[kf][j + 8] = (__bf16)wr[k0 + 16 + j];
            }
        }
    }

    const int   col  = wave * 16 + l15;            // C layout: N = l15 (per half)
    const float bias = Wb[col] + u_sh[col];
    const float vwc  = Vw[col];

    // output accumulators: thread -> (d = tid&127, row-half = tid>>7)
    const int dcol  = tid & 127;
    const int dhalf = tid >> 7;
    float num_acc   = 0.0f;
    float den_local = 0.0f;   // meaningful on tid<32 (stage-2 wave) only

    // staging map: thread owns 16 contiguous floats of the 32x128 tile
    const int srow = tid >> 3;          // 0..31
    const int scol = (tid & 7) * 16;    // 0..112

    // reduction stage-1 map: thread -> (row rm(+16), 8-col chunk cg)
    const int rm = tid >> 4;            // 0..15
    const int cg = tid & 15;            // 0..15

    const float* __restrict__ mrow = mask + (size_t)b * LSEQ;

    // ---- software pipeline: preload tile 0 into registers ----
    f4 r0, r1, r2, r3;
    {
        const float* src = Xb + srow * DIM + scol;
        r0 = *(const f4*)(src);
        r1 = *(const f4*)(src + 4);
        r2 = *(const f4*)(src + 8);
        r3 = *(const f4*)(src + 12);
    }

    int p = 0;
    for (int lt = 0; lt < LSEQ; lt += TILE_M, p ^= 1) {
        // drain staged registers into LDS buffer p (fp32 + bf16 pack)
        *(f4*)&x_sh[p][srow][scol]      = r0;
        *(f4*)&x_sh[p][srow][scol + 4]  = r1;
        *(f4*)&x_sh[p][srow][scol + 8]  = r2;
        *(f4*)&x_sh[p][srow][scol + 12] = r3;
        {
            v8bf pk0, pk1;
            #pragma unroll
            for (int j = 0; j < 4; ++j) {
                pk0[j] = (__bf16)r0[j]; pk0[j + 4] = (__bf16)r1[j];
                pk1[j] = (__bf16)r2[j]; pk1[j + 4] = (__bf16)r3[j];
            }
            *(v8bf*)&xb_sh[p][srow][scol]     = pk0;
            *(v8bf*)&xb_sh[p][srow][scol + 8] = pk1;
        }

        // issue NEXT tile's global loads now; consumed at next iteration's top,
        // so the whole compute phase below hides their latency
        if (lt + TILE_M < LSEQ) {
            const float* nsrc = Xb + (size_t)(lt + TILE_M) * DIM + srow * DIM + scol;
            r0 = *(const f4*)(nsrc);
            r1 = *(const f4*)(nsrc + 4);
            r2 = *(const f4*)(nsrc + 8);
            r3 = *(const f4*)(nsrc + 12);
        }
        if (lt + 2 * TILE_M < LSEQ)
            __builtin_prefetch(Xb + (size_t)(lt + 2 * TILE_M) * DIM + (tid << 4), 0, 1);
        __syncthreads();   // barrier 1: tile p visible

        // two 16-row sub-tiles, reusing resident bfrag: 8 WMMAs per iteration
        #pragma unroll
        for (int sub = 0; sub < NSUB; ++sub) {
            const int mbase = sub * 16;
            v16bf afrag[4];
            #pragma unroll
            for (int kf = 0; kf < 4; ++kf) {
                const int k0 = kf * 32 + lhalf * 8;
                v8bf lo = *(const v8bf*)&xb_sh[p][mbase + l15][k0];
                v8bf hi = *(const v8bf*)&xb_sh[p][mbase + l15][k0 + 16];
                afrag[kf] = __builtin_shufflevector(lo, hi,
                    0, 1, 2, 3, 4, 5, 6, 7, 8, 9, 10, 11, 12, 13, 14, 15);
            }
            v8f c = {};
            #pragma unroll
            for (int kf = 0; kf < 4; ++kf)
                c = __builtin_amdgcn_wmma_f32_16x16x32_bf16(
                        false, afrag[kf], false, bfrag[kf],
                        (short)0, c, false, false);
            // bias + tanh + Vw weight; scatter: C VGPR r -> row (r + 8*lhalf)
            #pragma unroll
            for (int r = 0; r < 8; ++r)
                part_sh[mbase + lhalf * 8 + r][col] = FAST_TANH(c[r] + bias) * vwc;
        }
        __syncthreads();   // barrier 2: part_sh complete

        // reduction stage 1: each thread sums two 8-col chunks
        #pragma unroll
        for (int s = 0; s < NSUB; ++s) {
            const int row = rm + 16 * s;
            f4 s0 = *(const f4*)&part_sh[row][cg * 8];
            f4 s1 = *(const f4*)&part_sh[row][cg * 8 + 4];
            f4 t  = s0 + s1;
            red_sh[row][cg] = (t.x + t.y) + (t.z + t.w);
        }
        __syncthreads();   // barrier 3: red_sh complete

        // reduction stage 2 (exactly one full wave) + e = exp(score)*mask
        if (tid < TILE_M) {
            f4 q0 = *(const f4*)&red_sh[tid][0];
            f4 q1 = *(const f4*)&red_sh[tid][4];
            f4 q2 = *(const f4*)&red_sh[tid][8];
            f4 q3 = *(const f4*)&red_sh[tid][12];
            f4 t  = (q0 + q1) + (q2 + q3);
            const float score = (t.x + t.y) + (t.z + t.w);
            const float e = __expf(score) * mrow[lt + tid];
            e_sh[tid] = e;
            den_local += e;    // running partition function, per-row slice
        }
        __syncthreads();   // barrier 4: e_sh visible

        // running weighted sum (single pass over L); double-buffered tiles
        // make a trailing barrier unnecessary
        #pragma unroll
        for (int r = 0; r < 16; ++r) {
            const int m = dhalf * 16 + r;
            num_acc = fmaf(e_sh[m], x_sh[p][m][dcol], num_acc);
        }
    }

    // ---- epilogue: combine numerator halves and the 32 den partials ----
    __syncthreads();                        // last-iteration e_sh readers done
    num_sh[dhalf][dcol] = num_acc;
    if (tid < TILE_M) e_sh[tid] = den_local;
    __syncthreads();
    if (tid < DIM) {
        float den = 0.0f;
        #pragma unroll
        for (int i = 0; i < TILE_M; ++i) den += e_sh[i];
        out[(size_t)b * DIM + tid] = (num_sh[0][tid] + num_sh[1][tid]) / den;
    }
}

extern "C" void kernel_launch(void* const* d_in, const int* in_sizes, int n_in,
                              void* d_out, int out_size, void* d_ws, size_t ws_size,
                              hipStream_t stream) {
    (void)in_sizes; (void)n_in; (void)d_ws; (void)ws_size; (void)out_size;
    const float* inputs = (const float*)d_in[0];
    const float* query  = (const float*)d_in[1];
    const float* mask   = (const float*)d_in[2];
    const float* Ww     = (const float*)d_in[3];
    const float* Wb     = (const float*)d_in[4];
    const float* Uw     = (const float*)d_in[5];
    const float* Ub     = (const float*)d_in[6];
    const float* Vw     = (const float*)d_in[7];
    float* out          = (float*)d_out;

    attn_additive_kernel<<<dim3(BATCH), dim3(256), 0, stream>>>(
        inputs, query, mask, Ww, Wb, Uw, Ub, Vw, out);
}
